// CrossDomainFusion_7756710937377
// MI455X (gfx1250) — matrix operand order repaired and checked by reading, not compiled, not run on hardware
//
#include <hip/hip_runtime.h>
#include <cstdint>
#include <cstddef>

// ---------------- types ----------------
typedef __attribute__((ext_vector_type(16))) __bf16 v16bf;
typedef __attribute__((ext_vector_type(8)))  float  v8f;
typedef __attribute__((ext_vector_type(4)))  unsigned int u32x4;

struct bf16x16 { u32x4 lo, hi; };
static_assert(sizeof(bf16x16) == 32, "frag size");

__device__ __forceinline__ v16bf ld_frag2(const __bf16* p0, const __bf16* p1) {
  bf16x16 t;
  t.lo = *(const u32x4*)p0;
  t.hi = *(const u32x4*)p1;
  return __builtin_bit_cast(v16bf, t);
}
__device__ __forceinline__ v16bf ld_frag1(const __bf16* p) {
  bf16x16 t;
  t.lo = *(const u32x4*)(p);
  t.hi = *(const u32x4*)(p + 8);
  return __builtin_bit_cast(v16bf, t);
}
__device__ __forceinline__ __bf16 f2bf(float f) {
  unsigned u = __builtin_bit_cast(unsigned, f);
  u += 0x7FFFu + ((u >> 16) & 1u);          // round-to-nearest-even
  unsigned short h = (unsigned short)(u >> 16);
  return __builtin_bit_cast(__bf16, h);
}
__device__ __forceinline__ float bf2f(__bf16 x) {
  unsigned short h = __builtin_bit_cast(unsigned short, x);
  unsigned u = ((unsigned)h) << 16;
  return __builtin_bit_cast(float, u);
}

// CDNA5 async global->LDS DMA path (ASYNCcnt), with portable fallback.
#if defined(__gfx1250__) && __has_builtin(__builtin_amdgcn_global_load_async_to_lds_b128) && __has_builtin(__builtin_amdgcn_s_wait_asynccnt)
#define HAVE_ASYNC_LDS 1
// Builtin parameter types per hipcc diagnostic: pointer to GCC-style int vec4,
// global (AS1) source and LDS (AS3) destination.
typedef int v4i_async __attribute__((vector_size(16)));
typedef __attribute__((address_space(1))) v4i_async* as1_v4i;
typedef __attribute__((address_space(3))) v4i_async* as3_v4i;
#else
#define HAVE_ASYNC_LDS 0
#endif

// ---------------- problem constants ----------------
#define BATCH   8
#define LTIME   1024
#define TDIM    256
#define TT      2048     // fused time length
#define CF      192      // C_SPEC * F_SPEC
#define HID     512

// ===================================================================
// Kernel 0: swizzle fp32 weight [N][K] -> bf16 B-fragment order
// tile(kt,nt): 32 lanes x 16 bf16; lane l: N = nt*16 + l%16,
//              K = kt*32 + (l/16)*16 + v  (v = 0..15 contiguous)
// ===================================================================
__global__ void prep_weight_kernel(const float* __restrict__ w,
                                   __bf16* __restrict__ out, int N, int K) {
  int idx = blockIdx.x * blockDim.x + threadIdx.x;
  int total = N * K;
  if (idx >= total) return;
  int tile   = idx >> 9;          // / 512
  int within = idx & 511;
  int lane   = within >> 4;
  int v      = within & 15;
  int ntiles = N >> 4;
  int kt = tile / ntiles;
  int nt = tile - kt * ntiles;
  int k = kt * 32 + (lane >> 4) * 16 + v;
  int n = nt * 16 + (lane & 15);
  out[idx] = f2bf(w[(size_t)n * K + k]);
}

// ===================================================================
// Kernel 1: conv_transpose1d (stride 2, k=4, pad 1) -> tf bf16 [B][TT][TDIM]
//   t even: x[t/2-1]*w[...,3] + x[t/2]*w[...,1]
//   t odd : x[(t-1)/2]*w[...,2] + x[(t+1)/2]*w[...,0]
// ===================================================================
__global__ void convt_kernel(const float* __restrict__ x,   // [B][LTIME][TDIM]
                             const float* __restrict__ w,   // [TDIM][TDIM][4] (in,out,k)
                             const float* __restrict__ bc,  // [TDIM]
                             __bf16* __restrict__ tf) {     // [B][TT][TDIM]
  int bt = blockIdx.x;
  int b = bt >> 11;
  int t = bt & (TT - 1);
  int o = threadIdx.x;                  // 0..255 output channel
  __shared__ float xs0[TDIM];
  __shared__ float xs1[TDIM];
  int r0, r1, k0, k1;
  if (t & 1) { r0 = (t - 1) >> 1; r1 = (t + 1) >> 1; k0 = 2; k1 = 0; }
  else       { r0 = (t >> 1) - 1; r1 = (t >> 1);     k0 = 3; k1 = 1; }
  bool v0 = (r0 >= 0) && (r0 < LTIME);
  bool v1 = (r1 < LTIME);
  xs0[o] = v0 ? x[((size_t)b * LTIME + r0) * TDIM + o] : 0.f;
  xs1[o] = v1 ? x[((size_t)b * LTIME + r1) * TDIM + o] : 0.f;
  __syncthreads();
  float acc = bc[o];
  #pragma unroll 4
  for (int i = 0; i < TDIM; ++i) {
    const float* wp = w + ((size_t)i * TDIM + o) * 4;
    acc = fmaf(xs0[i], wp[k0], acc);
    acc = fmaf(xs1[i], wp[k1], acc);
  }
  tf[((size_t)b * TT + t) * TDIM + o] = f2bf(acc);
}

// ===================================================================
// Kernel 2: spec (B,3,64,T) -> sf bf16 [B][TT][CF]
// ===================================================================
__global__ void spec_tr_kernel(const float* __restrict__ spec,
                               __bf16* __restrict__ sf) {
  int idx = blockIdx.x * blockDim.x + threadIdx.x;
  const int total = BATCH * TT * CF;
  if (idx >= total) return;
  int b  = idx / (TT * CF);
  int r  = idx - b * (TT * CF);
  int t  = r / CF;
  int cf = r - t * CF;
  int c  = cf >> 6;
  int f  = cf & 63;
  sf[idx] = f2bf(spec[(((size_t)b * 3 + c) * 64 + f) * TT + t]);
}

// ===================================================================
// Kernel 3: H = X(bf16 [M][K]) @ Wsw + bias  ->  Hb [M][512] bf16
//           and HbT [B][512][TT] bf16 (transposed per batch)
// Block = 256 threads = 8 waves; blockIdx.x = 16-row M tile.
// wave w covers N = w*64 .. w*64+63 (4 WMMA tiles), K-loop step 32.
// ===================================================================
__global__ void gemm_proj_kernel(const __bf16* __restrict__ X,
                                 const __bf16* __restrict__ Wsw,
                                 const float* __restrict__ bias,
                                 __bf16* __restrict__ Hb,
                                 __bf16* __restrict__ HbT,
                                 int K) {
  const int N = HID;
  const int ntiles = N >> 4;            // 32
  int m0   = blockIdx.x * 16;
  int lane = threadIdx.x & 31;
  int wv   = threadIdx.x >> 5;          // 0..7
  int lh   = lane >> 4;
  int lm   = lane & 15;

  v8f acc0 = {0.f,0.f,0.f,0.f,0.f,0.f,0.f,0.f};
  v8f acc1 = acc0, acc2 = acc0, acc3 = acc0;

  const __bf16* xrow = X + (size_t)(m0 + lm) * K;
  for (int k0 = 0; k0 < K; k0 += 32) {
    v16bf a = ld_frag2(xrow + k0 + lh * 16, xrow + k0 + lh * 16 + 8);
    int kt = k0 >> 5;
    const __bf16* wb =
        Wsw + ((size_t)(kt * ntiles + wv * 4) * 512) + (size_t)lane * 16;
    __builtin_prefetch(wb + 4 * 512, 0, 0);
    v16bf b0 = ld_frag1(wb);
    v16bf b1 = ld_frag1(wb + 512);
    v16bf b2 = ld_frag1(wb + 1024);
    v16bf b3 = ld_frag1(wb + 1536);
    acc0 = __builtin_amdgcn_wmma_f32_16x16x32_bf16(false, a, false, b0, (short)0, acc0, false, false);
    acc1 = __builtin_amdgcn_wmma_f32_16x16x32_bf16(false, a, false, b1, (short)0, acc1, false, false);
    acc2 = __builtin_amdgcn_wmma_f32_16x16x32_bf16(false, a, false, b2, (short)0, acc2, false, false);
    acc3 = __builtin_amdgcn_wmma_f32_16x16x32_bf16(false, a, false, b3, (short)0, acc3, false, false);
  }

  v8f accs[4] = {acc0, acc1, acc2, acc3};
  #pragma unroll
  for (int nt = 0; nt < 4; ++nt) {
    int col = wv * 64 + nt * 16 + lm;
    float bv = bias[col];
    #pragma unroll
    for (int r = 0; r < 8; ++r) {
      int row = m0 + r + 8 * lh;
      float vfl = accs[nt][r] + bv;
      __bf16 hb = f2bf(vfl);
      Hb[(size_t)row * HID + col] = hb;
      int bb = row >> 11;
      int tt = row & (TT - 1);
      HbT[((size_t)bb * HID + col) * TT + tt] = hb;
    }
  }
}

// ===================================================================
// Kernel 4: fused cross-attention for one direction.
//   Q = Qb rows [b][t][512], K = Kb, V given transposed VT [b][512][t].
//   Block handles 16 query rows; S (16x2048 f32) + P (bf16) in LDS.
//   out[b][t][dirOff + d] = softmax(Q K^T / sqrt(512)) @ V
// ===================================================================
__global__ void attn_kernel(const __bf16* __restrict__ Qb,
                            const __bf16* __restrict__ Kb,
                            const __bf16* __restrict__ VT,
                            float* __restrict__ out, int dirOff) {
  extern __shared__ char smem[];
  float*  S   = (float*)smem;                                // 16*2048 f32
  __bf16* P   = (__bf16*)(smem + 16 * TT * 4);               // 16*2048 bf16
  __bf16* Qs  = (__bf16*)(smem + 16 * TT * 4 + 16 * TT * 2); // 16*512 bf16
  float*  red = (float*)(smem + 16 * TT * 4 + 16 * TT * 2 + 16 * HID * 2);

  int m0 = blockIdx.x * 16;
  int b  = blockIdx.y;
  int tid  = threadIdx.x;
  int lane = tid & 31, wv = tid >> 5, lh = lane >> 4, lm = lane & 15;

  // ---- stage Q tile (16 rows x 512 bf16 = 16KB contiguous) into LDS ----
#if HAVE_ASYNC_LDS
  {
    char* src = (char*)(Qb + ((size_t)b * TT + m0) * HID);
    char* dst = (char*)Qs;
    #pragma unroll
    for (int it = 0; it < (16 * HID * 2) / (256 * 16); ++it) {
      int byteoff = (it * 256 + tid) * 16;
      __builtin_amdgcn_global_load_async_to_lds_b128(
          (as1_v4i)(src + byteoff), (as3_v4i)(dst + byteoff), 0, 0);
    }
    __builtin_amdgcn_s_wait_asynccnt(0);
  }
#else
  {
    const u32x4* src = (const u32x4*)(Qb + ((size_t)b * TT + m0) * HID);
    u32x4* dst = (u32x4*)Qs;
    for (int i = tid; i < (16 * HID * 2) / 16; i += 256) dst[i] = src[i];
  }
#endif
  __syncthreads();

  const float rs = 0.04419417382415922f; // 1/sqrt(512)

  // ---- S = Q K^T: each wave does 8 iterations of 2 j-tiles (ILP=2) ----
  for (int jt = 0; jt < 8; ++jt) {
    int j0 = wv * 256 + jt * 32;         // two adjacent 16-col tiles
    v8f accA = {0.f,0.f,0.f,0.f,0.f,0.f,0.f,0.f};
    v8f accB = accA;
    const __bf16* krowA = Kb + ((size_t)b * TT + j0 + lm) * HID + lh * 16;
    const __bf16* krowB = krowA + (size_t)16 * HID;
    const __bf16* qrow  = Qs + (size_t)lm * HID;
    #pragma unroll 4
    for (int k0 = 0; k0 < HID; k0 += 32) {
      v16bf a  = ld_frag2(qrow + k0 + lh * 16, qrow + k0 + lh * 16 + 8);
      v16bf bA = ld_frag1(krowA + k0);
      v16bf bB = ld_frag1(krowB + k0);
      accA = __builtin_amdgcn_wmma_f32_16x16x32_bf16(false, a, false, bA, (short)0, accA, false, false);
      accB = __builtin_amdgcn_wmma_f32_16x16x32_bf16(false, a, false, bB, (short)0, accB, false, false);
    }
    #pragma unroll
    for (int r = 0; r < 8; ++r) {
      S[(size_t)(r + 8 * lh) * TT + j0 + lm]      = accA[r] * rs;
      S[(size_t)(r + 8 * lh) * TT + j0 + 16 + lm] = accB[r] * rs;
    }
  }
  __syncthreads();

  // ---- softmax over each of the 16 rows (16 threads / row) ----
  int row = tid >> 4, seg = tid & 15;
  float* srow = S + (size_t)row * TT;
  __bf16* prow = P + (size_t)row * TT;
  float mx = -3.0e38f;
  for (int j = seg * 128; j < seg * 128 + 128; ++j) mx = fmaxf(mx, srow[j]);
  red[row * 16 + seg] = mx;
  __syncthreads();
  float m = -3.0e38f;
  #pragma unroll
  for (int s = 0; s < 16; ++s) m = fmaxf(m, red[row * 16 + s]);
  __syncthreads();
  float sum = 0.f;
  for (int j = seg * 128; j < seg * 128 + 128; ++j) {
    float e = __expf(srow[j] - m);
    prow[j] = f2bf(e);
    sum += e;
  }
  red[row * 16 + seg] = sum;
  __syncthreads();
  float tot = 0.f;
  #pragma unroll
  for (int s = 0; s < 16; ++s) tot += red[row * 16 + s];
  float inv = 1.0f / tot;
  for (int j = seg * 128; j < seg * 128 + 128; ++j)
    prow[j] = f2bf(bf2f(prow[j]) * inv);
  __syncthreads();

  // ---- O = P @ V  (wave covers 16 x 64 of output, 4 WMMA chains) ----
  v8f o0 = {0.f,0.f,0.f,0.f,0.f,0.f,0.f,0.f};
  v8f o1 = o0, o2 = o0, o3 = o0;
  const __bf16* parow = P + (size_t)lm * TT;
  const __bf16* vbase = VT + ((size_t)b * HID + wv * 64 + lm) * TT + lh * 16;
  for (int j0 = 0; j0 < TT; j0 += 32) {
    v16bf a = ld_frag2(parow + j0 + lh * 16, parow + j0 + lh * 16 + 8);
    __builtin_prefetch(vbase + j0 + 64, 0, 0);
    v16bf b0 = ld_frag1(vbase + j0);
    v16bf b1 = ld_frag1(vbase + (size_t)16 * TT + j0);
    v16bf b2 = ld_frag1(vbase + (size_t)32 * TT + j0);
    v16bf b3 = ld_frag1(vbase + (size_t)48 * TT + j0);
    o0 = __builtin_amdgcn_wmma_f32_16x16x32_bf16(false, a, false, b0, (short)0, o0, false, false);
    o1 = __builtin_amdgcn_wmma_f32_16x16x32_bf16(false, a, false, b1, (short)0, o1, false, false);
    o2 = __builtin_amdgcn_wmma_f32_16x16x32_bf16(false, a, false, b2, (short)0, o2, false, false);
    o3 = __builtin_amdgcn_wmma_f32_16x16x32_bf16(false, a, false, b3, (short)0, o3, false, false);
  }
  v8f oc[4] = {o0, o1, o2, o3};
  #pragma unroll
  for (int nt = 0; nt < 4; ++nt) {
    int col = wv * 64 + nt * 16 + lm;
    #pragma unroll
    for (int r = 0; r < 8; ++r) {
      int t = m0 + r + 8 * lh;
      out[((size_t)b * TT + t) * 1024 + dirOff + col] = oc[nt][r];
    }
  }
}

// ===================================================================
// host launcher
// ===================================================================
extern "C" void kernel_launch(void* const* d_in, const int* in_sizes, int n_in,
                              void* d_out, int out_size, void* d_ws, size_t ws_size,
                              hipStream_t stream) {
  (void)in_sizes; (void)n_in; (void)out_size; (void)ws_size;
  const float* time_f  = (const float*)d_in[0];
  const float* spec_f  = (const float*)d_in[1];
  const float* w_conv  = (const float*)d_in[2];
  const float* b_conv  = (const float*)d_in[3];
  const float* w_tproj = (const float*)d_in[4];
  const float* b_tproj = (const float*)d_in[5];
  const float* w_sproj = (const float*)d_in[6];
  const float* b_sproj = (const float*)d_in[7];
  float* out = (float*)d_out;

  // workspace carve-up (256B aligned)
  char* ws = (char*)d_ws;
  size_t off = 0;
  auto take = [&](size_t bytes) { char* p = ws + off; off = (off + bytes + 255) & ~(size_t)255; return p; };
  __bf16* tf    = (__bf16*)take((size_t)BATCH * TT * TDIM * 2);   // 8 MB
  __bf16* sf    = (__bf16*)take((size_t)BATCH * TT * CF * 2);     // 6 MB
  __bf16* WtSw  = (__bf16*)take((size_t)HID * TDIM * 2);
  __bf16* WsSw  = (__bf16*)take((size_t)HID * CF * 2);
  __bf16* Htb   = (__bf16*)take((size_t)BATCH * TT * HID * 2);    // 16 MB
  __bf16* Hsb   = (__bf16*)take((size_t)BATCH * TT * HID * 2);
  __bf16* HtTb  = (__bf16*)take((size_t)BATCH * HID * TT * 2);
  __bf16* HsTb  = (__bf16*)take((size_t)BATCH * HID * TT * 2);

  // 0) swizzle projection weights into WMMA B-fragment order
  {
    int tot_t = HID * TDIM;
    prep_weight_kernel<<<(tot_t + 255) / 256, 256, 0, stream>>>(w_tproj, WtSw, HID, TDIM);
    int tot_s = HID * CF;
    prep_weight_kernel<<<(tot_s + 255) / 256, 256, 0, stream>>>(w_sproj, WsSw, HID, CF);
  }
  // 1) conv-transpose -> tf bf16
  convt_kernel<<<BATCH * TT, TDIM, 0, stream>>>(time_f, w_conv, b_conv, tf);
  // 2) spec transpose -> sf bf16
  {
    int tot = BATCH * TT * CF;
    spec_tr_kernel<<<(tot + 255) / 256, 256, 0, stream>>>(spec_f, sf);
  }
  // 3) projections (WMMA)
  gemm_proj_kernel<<<(BATCH * TT) / 16, 256, 0, stream>>>(tf, WtSw, b_tproj, Htb, HtTb, TDIM);
  gemm_proj_kernel<<<(BATCH * TT) / 16, 256, 0, stream>>>(sf, WsSw, b_sproj, Hsb, HsTb, CF);
  // 4) bidirectional cross-attention (WMMA + LDS softmax)
  size_t smem = (size_t)16 * TT * 4 + (size_t)16 * TT * 2 + (size_t)16 * HID * 2 + 256 * 4;
  (void)hipFuncSetAttribute((const void*)attn_kernel,
                            hipFuncAttributeMaxDynamicSharedMemorySize, (int)smem);
  dim3 agrid(TT / 16, BATCH);
  attn_kernel<<<agrid, 256, smem, stream>>>(Htb, Hsb, HsTb, out, 0);
  attn_kernel<<<agrid, 256, smem, stream>>>(Hsb, Htb, HtTb, out, HID);
}
